// Attn_3075196583962
// MI455X (gfx1250) — matrix operand order repaired and checked by reading, not compiled, not run on hardware
//
#include <hip/hip_runtime.h>
#include <hip/hip_fp16.h>

typedef __attribute__((ext_vector_type(16))) _Float16 v16h;
typedef __attribute__((ext_vector_type(8)))  _Float16 v8h;
typedef __attribute__((ext_vector_type(8)))  float    v8f;

#define N_Q 1024
#define M_K 1024
#define D_F 512
#define ROWS 4            // query rows per score block

#if __has_builtin(__builtin_amdgcn_tanhf)
__device__ __forceinline__ float fast_tanh(float x) { return __builtin_amdgcn_tanhf(x); }
#elif __has_builtin(__builtin_amdgcn_tanh_f32)
__device__ __forceinline__ float fast_tanh(float x) { return __builtin_amdgcn_tanh_f32(x); }
#else
__device__ __forceinline__ float fast_tanh(float x) { return tanhf(x); }
#endif

// ---------------------------------------------------------------- f32 -> f16
struct H4 { _Float16 a, b, c, d; };

__global__ __launch_bounds__(256) void cvt_f16(const float* __restrict__ in,
                                               _Float16* __restrict__ out, int n4) {
    int i = blockIdx.x * 256 + threadIdx.x;
    if (i < n4) {
        float4 f = ((const float4*)in)[i];
        H4 h; h.a = (_Float16)f.x; h.b = (_Float16)f.y;
        h.c = (_Float16)f.z; h.d = (_Float16)f.w;
        ((H4*)out)[i] = h;
    }
}

// --------------------------------------------- transpose f32[R,C] -> f16[C,R]
__global__ __launch_bounds__(256) void transpose_to_f16(const float* __restrict__ in,
                                                        _Float16* __restrict__ out,
                                                        int R, int C) {
    __shared__ float tile[32][33];
    int c0 = blockIdx.x * 32, r0 = blockIdx.y * 32;
    int tx = threadIdx.x, ty = threadIdx.y;            // block (32, 8)
    #pragma unroll
    for (int i = 0; i < 32; i += 8)
        tile[ty + i][tx] = in[(size_t)(r0 + ty + i) * C + c0 + tx];
    __syncthreads();
    #pragma unroll
    for (int i = 0; i < 32; i += 8)
        out[(size_t)(c0 + ty + i) * R + r0 + tx] = (_Float16)tile[tx][ty + i];
}

// -------------------------------------------------- WMMA fragment loaders
__device__ __forceinline__ v16h load_a_frag(const _Float16* base, int lda, int lane) {
    int r = lane & 15, h = lane >> 4;
    const _Float16* p = base + (size_t)r * lda + h * 8;
    v8h lo = *(const v8h*)(p);
    v8h hi = *(const v8h*)(p + 16);
    return __builtin_shufflevector(lo, hi, 0,1,2,3,4,5,6,7,8,9,10,11,12,13,14,15);
}

__device__ __forceinline__ v16h load_b_frag(const _Float16* base, int ldb, int lane) {
    int c = lane & 15, koff = (lane >> 4) * 8;
    const _Float16* p = base + (size_t)c * ldb + koff;
    v8h lo = *(const v8h*)(p);
    v8h hi = *(const v8h*)(p + 16);
    return __builtin_shufflevector(lo, hi, 0,1,2,3,4,5,6,7,8,9,10,11,12,13,14,15);
}

// ---------------------------------------------------------------- WMMA GEMM
// C[row,col] = sum_k A[row,k] * Bcol[col][k] (+ bias[col]); C f32 row-major.
__global__ __launch_bounds__(128) void wmma_gemm_f16(
    const _Float16* __restrict__ A, int lda,
    const _Float16* __restrict__ Bc, int ldb,
    const float* __restrict__ bias,
    float* __restrict__ C, int ldc, int K)
{
    int lane = threadIdx.x & 31;
    int wv   = threadIdx.x >> 5;
    int row0 = blockIdx.x * 64 + (wv >> 1) * 32;
    int col0 = blockIdx.y * 64 + (wv & 1) * 32;

    v8f c00 = {}, c01 = {}, c10 = {}, c11 = {};
    for (int k0 = 0; k0 < K; k0 += 32) {
        v16h a0 = load_a_frag(A + (size_t)row0 * lda + k0, lda, lane);
        v16h a1 = load_a_frag(A + (size_t)(row0 + 16) * lda + k0, lda, lane);
        v16h b0 = load_b_frag(Bc + (size_t)col0 * ldb + k0, ldb, lane);
        v16h b1 = load_b_frag(Bc + (size_t)(col0 + 16) * ldb + k0, ldb, lane);
        c00 = __builtin_amdgcn_wmma_f32_16x16x32_f16(false, a0, false, b0, (short)0, c00, false, false);
        c01 = __builtin_amdgcn_wmma_f32_16x16x32_f16(false, a0, false, b1, (short)0, c01, false, false);
        c10 = __builtin_amdgcn_wmma_f32_16x16x32_f16(false, a1, false, b0, (short)0, c10, false, false);
        c11 = __builtin_amdgcn_wmma_f32_16x16x32_f16(false, a1, false, b1, (short)0, c11, false, false);
    }
    int cc = lane & 15;
    int rh = (lane >> 4) * 8;
    float bv0 = bias ? bias[col0 + cc]      : 0.f;
    float bv1 = bias ? bias[col0 + 16 + cc] : 0.f;
    #pragma unroll
    for (int r = 0; r < 8; ++r) {
        int row = row0 + r + rh;
        C[(size_t)row * ldc + col0 + cc]             = c00[r] + bv0;
        C[(size_t)row * ldc + col0 + 16 + cc]        = c01[r] + bv1;
        C[(size_t)(row + 16) * ldc + col0 + cc]      = c10[r] + bv0;
        C[(size_t)(row + 16) * ldc + col0 + 16 + cc] = c11[r] + bv1;
    }
}

// ------------------------------------- Bahdanau scores + masked softmax
// Block = 256 thr (8 waves) owns ROWS=4 query rows. qp rows staged in LDS via
// global_load_async_to_lds_b128 (ASYNCcnt path). Each wave loads a kp row into
// registers ONCE and reuses it for all 4 query rows (4x less L2 traffic).
// Softmax: wave w (<4) owns row w entirely (shfl-only reductions).
__global__ __launch_bounds__(256) void scores_softmax(
    const float* __restrict__ qp, const float* __restrict__ kp,
    const float* __restrict__ Ww, const float* __restrict__ bw,
    const int* __restrict__ mask, _Float16* __restrict__ attn)
{
    __shared__ float qp8[ROWS][D_F];     //  8 KB
    __shared__ float sc[ROWS][M_K];      // 16 KB

    int tid = threadIdx.x, lane = tid & 31, w = tid >> 5;
    int n0 = blockIdx.x * ROWS;

    // ---- stage ROWS qp rows into LDS with async copies (ASYNCcnt) ----
    {
        const float* gsrc = qp + (size_t)n0 * D_F;
        unsigned lbase = (unsigned)(size_t)(&qp8[0][0]);
        #pragma unroll
        for (int r = 0; r < (ROWS * D_F) / (256 * 4); ++r) {   // 2 iters
            int t = tid + 256 * r;                             // 16B chunk id
            unsigned loff = lbase + (unsigned)t * 16u;
            const float* g = gsrc + t * 4;
            asm volatile("global_load_async_to_lds_b128 %0, %1, off"
                         :: "v"(loff), "v"(g) : "memory");
        }
    }
    // Ww into registers (per wave, lane-sliced)
    float4 wf0 = *(const float4*)(Ww + 0   + lane * 4);
    float4 wf1 = *(const float4*)(Ww + 128 + lane * 4);
    float4 wf2 = *(const float4*)(Ww + 256 + lane * 4);
    float4 wf3 = *(const float4*)(Ww + 384 + lane * 4);

    asm volatile("s_wait_asynccnt 0" ::: "memory");
    __syncthreads();

    // ---- phase 2: raw scores ----
    for (int mm = 0; mm < M_K / 8; ++mm) {
        int m = w * (M_K / 8) + mm;
        const float* krow = kp + (size_t)m * D_F + lane * 4;
        float4 kf0 = *(const float4*)(krow);
        float4 kf1 = *(const float4*)(krow + 128);
        float4 kf2 = *(const float4*)(krow + 256);
        float4 kf3 = *(const float4*)(krow + 384);
        if (mm + 1 < M_K / 8) __builtin_prefetch(krow + D_F, 0, 0);
        #pragma unroll
        for (int n = 0; n < ROWS; ++n) {
            const float* q4 = &qp8[n][lane * 4];
            float4 qf0 = *(const float4*)(q4);
            float4 qf1 = *(const float4*)(q4 + 128);
            float4 qf2 = *(const float4*)(q4 + 256);
            float4 qf3 = *(const float4*)(q4 + 384);
            float s;
            s  = wf0.x * fast_tanh(qf0.x + kf0.x);
            s += wf0.y * fast_tanh(qf0.y + kf0.y);
            s += wf0.z * fast_tanh(qf0.z + kf0.z);
            s += wf0.w * fast_tanh(qf0.w + kf0.w);
            s += wf1.x * fast_tanh(qf1.x + kf1.x);
            s += wf1.y * fast_tanh(qf1.y + kf1.y);
            s += wf1.z * fast_tanh(qf1.z + kf1.z);
            s += wf1.w * fast_tanh(qf1.w + kf1.w);
            s += wf2.x * fast_tanh(qf2.x + kf2.x);
            s += wf2.y * fast_tanh(qf2.y + kf2.y);
            s += wf2.z * fast_tanh(qf2.z + kf2.z);
            s += wf2.w * fast_tanh(qf2.w + kf2.w);
            s += wf3.x * fast_tanh(qf3.x + kf3.x);
            s += wf3.y * fast_tanh(qf3.y + kf3.y);
            s += wf3.z * fast_tanh(qf3.z + kf3.z);
            s += wf3.w * fast_tanh(qf3.w + kf3.w);
            #pragma unroll
            for (int off = 16; off >= 1; off >>= 1) s += __shfl_xor(s, off, 32);
            if (lane == 0) sc[n][m] = s;
        }
    }
    __syncthreads();

    // ---- phase 3: per-wave masked softmax; wave w owns row n0+w ----
    if (w < ROWS) {
        int n = n0 + w;
        float bwv = bw[0];
        float mx = -3.4e38f;
        #pragma unroll
        for (int j = 0; j < M_K / 32; ++j) {
            int m = j * 32 + lane;
            float s = mask[(size_t)n * M_K + m] ? (sc[w][m] + bwv) : -1000000.0f;
            sc[w][m] = s;
            mx = fmaxf(mx, s);
        }
        #pragma unroll
        for (int off = 16; off >= 1; off >>= 1) mx = fmaxf(mx, __shfl_xor(mx, off, 32));
        float sum = 0.f;
        #pragma unroll
        for (int j = 0; j < M_K / 32; ++j) {
            int m = j * 32 + lane;
            float e = __expf(sc[w][m] - mx);
            sc[w][m] = e;
            sum += e;
        }
        #pragma unroll
        for (int off = 16; off >= 1; off >>= 1) sum += __shfl_xor(sum, off, 32);
        float inv = 1.f / sum;
        #pragma unroll
        for (int j = 0; j < M_K / 32; ++j) {
            int m = j * 32 + lane;
            attn[(size_t)n * M_K + m] = (_Float16)(sc[w][m] * inv);
        }
    }
}

// ---------------------------------------------------------------------------
extern "C" void kernel_launch(void* const* d_in, const int* in_sizes, int n_in,
                              void* d_out, int out_size, void* d_ws, size_t ws_size,
                              hipStream_t stream) {
    (void)in_sizes; (void)n_in; (void)out_size; (void)ws_size;
    const float* q  = (const float*)d_in[0];
    const float* k  = (const float*)d_in[1];
    const float* v  = (const float*)d_in[2];
    const int*   mk = (const int*)  d_in[3];
    const float* Wq = (const float*)d_in[4];
    const float* bq = (const float*)d_in[5];
    const float* Wk = (const float*)d_in[6];
    const float* bk = (const float*)d_in[7];
    const float* Wv = (const float*)d_in[8];
    const float* bv = (const float*)d_in[9];
    const float* Ww = (const float*)d_in[10];
    const float* bw = (const float*)d_in[11];

    char* ws = (char*)d_ws;
    _Float16* q16  = (_Float16*)(ws + 0);          //  1 MB
    _Float16* k16  = (_Float16*)(ws + 1048576);    //  1 MB
    _Float16* v16  = (_Float16*)(ws + 2097152);    //  1 MB
    _Float16* Wq16 = (_Float16*)(ws + 3145728);    // 512 KB
    _Float16* Wk16 = (_Float16*)(ws + 3670016);    // 512 KB
    _Float16* Wv16 = (_Float16*)(ws + 4194304);    // 512 KB
    float*    qp   = (float*)   (ws + 4718592);    //  2 MB
    float*    kp   = (float*)   (ws + 6815744);    //  2 MB
    float*    vp   = (float*)   (ws + 8912896);    //  2 MB
    _Float16* vpT  = (_Float16*)(ws + 11010048);   //  1 MB  [D x M]
    _Float16* attn = (_Float16*)(ws + 12058624);   //  2 MB  [N x M]

    // 1) f32 -> f16 casts
    cvt_f16<<<(N_Q * D_F / 4 + 255) / 256, 256, 0, stream>>>(q,  q16,  N_Q * D_F / 4);
    cvt_f16<<<(M_K * D_F / 4 + 255) / 256, 256, 0, stream>>>(k,  k16,  M_K * D_F / 4);
    cvt_f16<<<(M_K * D_F / 4 + 255) / 256, 256, 0, stream>>>(v,  v16,  M_K * D_F / 4);
    cvt_f16<<<(D_F * D_F / 4 + 255) / 256, 256, 0, stream>>>(Wq, Wq16, D_F * D_F / 4);
    cvt_f16<<<(D_F * D_F / 4 + 255) / 256, 256, 0, stream>>>(Wk, Wk16, D_F * D_F / 4);
    cvt_f16<<<(D_F * D_F / 4 + 255) / 256, 256, 0, stream>>>(Wv, Wv16, D_F * D_F / 4);

    // 2) projections (WMMA): out[i,d] = sum_j X[i,j]*W[d,j] + b[d]
    dim3 gProj(N_Q / 64, D_F / 64);
    wmma_gemm_f16<<<gProj, 128, 0, stream>>>(q16, D_F, Wq16, D_F, bq, qp, D_F, D_F);
    wmma_gemm_f16<<<gProj, 128, 0, stream>>>(k16, D_F, Wk16, D_F, bk, kp, D_F, D_F);
    wmma_gemm_f16<<<gProj, 128, 0, stream>>>(v16, D_F, Wv16, D_F, bv, vp, D_F, D_F);

    // 3) vp [M,D] f32 -> vpT [D,M] f16
    transpose_to_f16<<<dim3(D_F / 32, M_K / 32), dim3(32, 8), 0, stream>>>(vp, vpT, M_K, D_F);

    // 4) additive scores + masked softmax (trans-pipe bound hot loop)
    scores_softmax<<<N_Q / ROWS, 256, 0, stream>>>(qp, kp, Ww, bw, mk, attn);

    // 5) context = attn @ vp (WMMA)
    wmma_gemm_f16<<<dim3(N_Q / 64, D_F / 64), 128, 0, stream>>>(
        attn, M_K, vpT, M_K, nullptr, (float*)d_out, D_F, M_K);
}
